// Head_8942121910703
// MI455X (gfx1250) — matrix operand order repaired and checked by reading, not compiled, not run on hardware
//
#include <hip/hip_runtime.h>
#include <hip/hip_bf16.h>
#include <stdint.h>

#define Bn 32
#define Tn 2048
#define Cn 512
#define Hn 64

typedef __bf16 bf16_t;
typedef __attribute__((ext_vector_type(16))) __bf16 v16bf;
typedef __attribute__((ext_vector_type(8)))  __bf16 v8bf;
typedef __attribute__((ext_vector_type(8)))  float  v8f;

union V16U { v16bf v; v8bf h[2]; };

__device__ __forceinline__ v8f wmma_bf16(v16bf a, v16bf b, v8f c) {
    return __builtin_amdgcn_wmma_f32_16x16x32_bf16(false, a, false, b, (short)0, c, false, false);
}

// ---------------------------------------------------------------------------
// Stage 0: W [C][H] fp32  ->  Wt [H][C] bf16  (so B-operand loads are contiguous)
// ---------------------------------------------------------------------------
__global__ __launch_bounds__(256) void transpose_w_kernel(const float* __restrict__ W,
                                                          bf16_t* __restrict__ Wt) {
    int idx = blockIdx.x * 256 + threadIdx.x;      // over C*H = 32768
    if (idx >= Cn * Hn) return;
    int c = idx / Hn;
    int h = idx % Hn;
    Wt[h * Cn + c] = (__bf16)W[idx];
}

// ---------------------------------------------------------------------------
// Stage 1: fused Q/K/V projection with v_wmma_f32_16x16x32_bf16.
// Block = 128 threads = 4 waves; wave w owns N-tile w (cols 16w..16w+15) for all of Q,K,V.
// Q,K written row-major [B*T][64] bf16; V written transposed Vt[b][h][t] bf16.
// ---------------------------------------------------------------------------
__global__ __launch_bounds__(128) void proj_qkv_kernel(const float* __restrict__ x,
                                                       const bf16_t* __restrict__ WqT,
                                                       const bf16_t* __restrict__ WkT,
                                                       const bf16_t* __restrict__ WvT,
                                                       bf16_t* __restrict__ Q,
                                                       bf16_t* __restrict__ K,
                                                       bf16_t* __restrict__ Vt) {
    const int wave = threadIdx.x >> 5;   // 0..3 = N tile
    const int lane = threadIdx.x & 31;
    const int m    = lane & 15;
    const int hi   = lane >> 4;          // 0/1
    const int row0 = blockIdx.x * 16;    // M tile over B*T
    const int row  = row0 + m;

    const float*  xrow = x + (size_t)row * Cn;
    const bf16_t* bq   = WqT + (size_t)(wave * 16 + m) * Cn + hi * 16;
    const bf16_t* bk   = WkT + (size_t)(wave * 16 + m) * Cn + hi * 16;
    const bf16_t* bv   = WvT + (size_t)(wave * 16 + m) * Cn + hi * 16;

    v8f accq = {}, acck = {}, accv = {};

    for (int kc = 0; kc < Cn; kc += 32) {
        // A operand (x tile 16x32, contiguous k order, same as B):
        // lane<16: K = kc+0..15 ; lane>=16: K = kc+16..31
        const float4* pa = (const float4*)(xrow + kc + hi * 16);
        float4 f0 = pa[0];
        float4 f1 = pa[1];
        const float4* pb = (const float4*)(xrow + kc + hi * 16 + 8);
        float4 f2 = pb[0];
        float4 f3 = pb[1];
        v16bf a;
        a[0]  = (__bf16)f0.x; a[1]  = (__bf16)f0.y; a[2]  = (__bf16)f0.z; a[3]  = (__bf16)f0.w;
        a[4]  = (__bf16)f1.x; a[5]  = (__bf16)f1.y; a[6]  = (__bf16)f1.z; a[7]  = (__bf16)f1.w;
        a[8]  = (__bf16)f2.x; a[9]  = (__bf16)f2.y; a[10] = (__bf16)f2.z; a[11] = (__bf16)f2.w;
        a[12] = (__bf16)f3.x; a[13] = (__bf16)f3.y; a[14] = (__bf16)f3.z; a[15] = (__bf16)f3.w;

        // B operands: 16 contiguous bf16 per lane (N = lane&15, K = kc + hi*16 + j)
        v16bf bqv = *(const v16bf*)(bq + kc);
        v16bf bkv = *(const v16bf*)(bk + kc);
        v16bf bvv = *(const v16bf*)(bv + kc);

        accq = wmma_bf16(a, bqv, accq);
        acck = wmma_bf16(a, bkv, acck);
        accv = wmma_bf16(a, bvv, accv);
    }

    // C/D layout: element r -> row M = r + 8*hi, col N = lane&15
    const int h = wave * 16 + m;
#pragma unroll
    for (int r = 0; r < 8; ++r) {
        int orow = row0 + r + hi * 8;       // global row in B*T
        Q[(size_t)orow * Hn + h] = (__bf16)accq[r];
        K[(size_t)orow * Hn + h] = (__bf16)acck[r];
        int b = orow >> 11;                 // / T
        int t = orow & (Tn - 1);
        Vt[((size_t)b * Hn + h) * Tn + t] = (__bf16)accv[r];
    }
}

// ---------------------------------------------------------------------------
// Stage 2: flash attention. Block = 256 threads = 8 waves; each wave owns a
// 16-row query tile. Key blocks of 32: 4 WMMAs for S, online softmax in fp32,
// P transposed via per-wave LDS tile, 4 WMMAs for P@V.
// ---------------------------------------------------------------------------
__global__ __launch_bounds__(256) void attn_kernel(const bf16_t* __restrict__ Q,
                                                   const bf16_t* __restrict__ K,
                                                   const bf16_t* __restrict__ Vt,
                                                   const uint8_t* __restrict__ mask,
                                                   float* __restrict__ out) {
    __shared__ bf16_t pshm[8][16 * 32];      // per-wave P tile (16 rows x 32 keys)

    const int wave  = threadIdx.x >> 5;
    const int lane  = threadIdx.x & 31;
    const int m     = lane & 15;
    const int hi    = lane >> 4;
    const int off0  = hi * 16;
    const int b     = blockIdx.y;
    const int qt    = blockIdx.x * 8 + wave; // query tile 0..127
    const int qrow0 = qt * 16;

    // ---- load Q A-operands for both H chunks (contiguous 16B pieces) ----
    const bf16_t* qrow = Q + ((size_t)b * Tn + qrow0 + m) * Hn;
    V16U aq0, aq1;
    aq0.h[0] = *(const v8bf*)(qrow + off0);
    aq0.h[1] = *(const v8bf*)(qrow + off0 + 8);
    aq1.h[0] = *(const v8bf*)(qrow + 32 + off0);
    aq1.h[1] = *(const v8bf*)(qrow + 32 + off0 + 8);

    const bf16_t*  Kb = K   + (size_t)b * Tn * Hn;
    const bf16_t*  Vb = Vt  + (size_t)b * Hn * Tn;
    const uint8_t* mb = mask + (size_t)b * Tn;
    bf16_t* pt = pshm[wave];

    float mrun[8], lrun[8];
#pragma unroll
    for (int r = 0; r < 8; ++r) { mrun[r] = -3.0e38f; lrun[r] = 0.0f; }
    v8f o0 = {}, o1 = {}, o2 = {}, o3 = {};

    const float sc   = 0.125f;               // 1/sqrt(H)
    const int   send = qrow0 + 16;           // causal: keys in [0, send)

    for (int s0 = 0; s0 < send; s0 += 32) {
        const bf16_t* krlo = Kb + (size_t)(s0 + m) * Hn + hi * 16;
        const bf16_t* krhi = Kb + (size_t)(s0 + 16 + m) * Hn + hi * 16;
        if (s0 + 32 < send) __builtin_prefetch(krlo + 32 * Hn, 0, 1);

        // ---- S = Q K^T (f32 accum), two 16-key N tiles, two H chunks ----
        v8f slo = {}, shi = {};
        slo = wmma_bf16(aq0.v, *(const v16bf*)(krlo),      slo);
        slo = wmma_bf16(aq1.v, *(const v16bf*)(krlo + 32), slo);
        shi = wmma_bf16(aq0.v, *(const v16bf*)(krhi),      shi);
        shi = wmma_bf16(aq1.v, *(const v16bf*)(krhi + 32), shi);

        // ---- masking (causal + padding) ----
        const int slo_idx = s0 + m;
        const int shi_idx = s0 + 16 + m;
        const int mlo = mb[slo_idx];
        const int mhi = (shi_idx < Tn) ? mb[shi_idx] : 0;

        float vlo[8], vhi[8];
#pragma unroll
        for (int r = 0; r < 8; ++r) {
            int qr = qrow0 + r + hi * 8;
            vlo[r] = ((slo_idx <= qr) && mlo) ? slo[r] * sc : -1.0e30f;
            vhi[r] = ((shi_idx <= qr) && mhi) ? shi[r] * sc : -1.0e30f;
        }

        // ---- online softmax: row reductions across each 16-lane half ----
#pragma unroll
        for (int r = 0; r < 8; ++r) {
            float mx = fmaxf(vlo[r], vhi[r]);
#pragma unroll
            for (int d = 1; d < 16; d <<= 1) mx = fmaxf(mx, __shfl_xor(mx, d, 32));
            float mnew  = fmaxf(mrun[r], mx);
            float alpha = __expf(mrun[r] - mnew);
            float plo   = __expf(vlo[r] - mnew);
            float phi   = __expf(vhi[r] - mnew);
            float rs    = plo + phi;
#pragma unroll
            for (int d = 1; d < 16; d <<= 1) rs += __shfl_xor(rs, d, 32);
            lrun[r] = lrun[r] * alpha + rs;
            mrun[r] = mnew;
            o0[r] *= alpha; o1[r] *= alpha; o2[r] *= alpha; o3[r] *= alpha;
            vlo[r] = plo; vhi[r] = phi;
        }

        // ---- transpose P (C/D layout -> A layout) through per-wave LDS ----
#pragma unroll
        for (int r = 0; r < 8; ++r) {
            int prow = r + hi * 8;
            pt[prow * 32 + m]      = (__bf16)vlo[r];
            pt[prow * 32 + 16 + m] = (__bf16)vhi[r];
        }
        // DS ops of one wave are processed in order -> no barrier needed.
        V16U ap;
        ap.h[0] = *(const v8bf*)(pt + m * 32 + off0);
        ap.h[1] = *(const v8bf*)(pt + m * 32 + off0 + 8);

        // ---- O += P V : B operand contiguous in key dim (V transposed) ----
        const bf16_t* vcol = Vb + (size_t)m * Tn + s0 + hi * 16;
        o0 = wmma_bf16(ap.v, *(const v16bf*)(vcol),            o0);
        o1 = wmma_bf16(ap.v, *(const v16bf*)(vcol + 16 * Tn), o1);
        o2 = wmma_bf16(ap.v, *(const v16bf*)(vcol + 32 * Tn), o2);
        o3 = wmma_bf16(ap.v, *(const v16bf*)(vcol + 48 * Tn), o3);
    }

    // ---- finalize: divide by row sums, store fp32 [B][T][H] ----
    float* ob = out + (size_t)b * Tn * Hn;
#pragma unroll
    for (int r = 0; r < 8; ++r) {
        float inv = 1.0f / lrun[r];
        int t = qrow0 + r + hi * 8;
        float* po = ob + (size_t)t * Hn + m;
        po[0]  = o0[r] * inv;
        po[16] = o1[r] * inv;
        po[32] = o2[r] * inv;
        po[48] = o3[r] * inv;
    }
}

// ---------------------------------------------------------------------------
extern "C" void kernel_launch(void* const* d_in, const int* in_sizes, int n_in,
                              void* d_out, int out_size, void* d_ws, size_t ws_size,
                              hipStream_t stream) {
    const float*   x  = (const float*)d_in[0];
    const float*   Wk = (const float*)d_in[1];
    const float*   Wq = (const float*)d_in[2];
    const float*   Wv = (const float*)d_in[3];
    const uint8_t* mk = (const uint8_t*)d_in[4];
    float* out = (float*)d_out;

    // workspace layout (bf16), 256B aligned chunks
    char* ws = (char*)d_ws;
    auto take = [&](size_t bytes) { char* p = ws; ws += (bytes + 255) & ~size_t(255); return p; };
    bf16_t* WqT = (bf16_t*)take((size_t)Hn * Cn * 2);
    bf16_t* WkT = (bf16_t*)take((size_t)Hn * Cn * 2);
    bf16_t* WvT = (bf16_t*)take((size_t)Hn * Cn * 2);
    bf16_t* Qb  = (bf16_t*)take((size_t)Bn * Tn * Hn * 2);
    bf16_t* Kb  = (bf16_t*)take((size_t)Bn * Tn * Hn * 2);
    bf16_t* Vt  = (bf16_t*)take((size_t)Bn * Hn * Tn * 2);

    const int twgrid = (Cn * Hn + 255) / 256;
    transpose_w_kernel<<<twgrid, 256, 0, stream>>>(Wq, WqT);
    transpose_w_kernel<<<twgrid, 256, 0, stream>>>(Wk, WkT);
    transpose_w_kernel<<<twgrid, 256, 0, stream>>>(Wv, WvT);

    proj_qkv_kernel<<<(Bn * Tn) / 16, 128, 0, stream>>>(x, WqT, WkT, WvT, Qb, Kb, Vt);

    dim3 agrid(Tn / (16 * 8), Bn);  // (16, 32)
    attn_kernel<<<agrid, 256, 0, stream>>>(Qb, Kb, Vt, mk, out);
}
